// AttentionMultiHead_9036611190885
// MI455X (gfx1250) — matrix operand order repaired and checked by reading, not compiled, not run on hardware
//
#include <hip/hip_runtime.h>
#include <hip/hip_bf16.h>

typedef _Float16 h16;
typedef __attribute__((ext_vector_type(16))) _Float16 v16h;
typedef __attribute__((ext_vector_type(8)))  float    v8f;
typedef __attribute__((ext_vector_type(4)))  unsigned int u32x4;
typedef __attribute__((ext_vector_type(8)))  int          i32x8;
typedef __attribute__((ext_vector_type(4)))  int          i32x4;

#define D_MODEL 1024
#define DK      64
#define NHEADS  16
#define BATCH   8
#define SEQ     1024
#define ROWS    (BATCH * SEQ)     /* 8192 */
#define BHN     (BATCH * NHEADS)  /* 128  */

// ---------------------------------------------------------------------------
// Fragment loaders per CDNA5 WMMA VGPR layouts (CDNA5 ISA §7.12.2)
// ---------------------------------------------------------------------------

// A-fragment 16x32 f16. base points at (row0, k0); ld = row stride (elements).
template <typename T>
static __device__ __forceinline__ v16h load_afrag(const T* base, int ld) {
  const int lane = threadIdx.x & 31;
  const int m    = lane & 15;
  const int hh   = lane >> 4;
  const T* p = base + m * ld + 8 * hh;
  v16h a;
#pragma unroll
  for (int i = 0; i < 8; ++i) {
    a[i]     = (_Float16)p[i];
    a[8 + i] = (_Float16)p[16 + i];
  }
  return a;
}

// B-fragment 32x16 where B[k][n] = src[(n0+n)*ld + (k0+k)]  (column-major /
// transposed source: k contiguous per column). base points at (n0, k0).
static __device__ __forceinline__ v16h load_bfrag_T(const h16* base, int ld) {
  const int lane = threadIdx.x & 31;
  const int n  = lane & 15;
  const int kb = (lane >> 4) * 8;
  const h16* p = base + n * ld + kb;
  v16h b;
#pragma unroll
  for (int i = 0; i < 8; ++i) {
    b[i]     = p[i];
    b[8 + i] = p[16 + i];
  }
  return b;
}

static __device__ __forceinline__ v8f wmma_f32(v16h a, v16h b, v8f c) {
  return __builtin_amdgcn_wmma_f32_16x16x32_f16(false, a, false, b, (short)0, c,
                                                false, false);
}

// ---------------------------------------------------------------------------
// Kernel 0 (one-time prep): convert+transpose all weights to f16 column-major
// Wq/Wk/Wv [16][1024][64] -> [16][64][1024];  Wo [1024][64] -> [64][1024]
// ---------------------------------------------------------------------------
__global__ __launch_bounds__(256) void prep_weights_kernel(
    const float* __restrict__ Wq, const float* __restrict__ Wk,
    const float* __restrict__ Wv, const float* __restrict__ Wo,
    h16* __restrict__ wqT, h16* __restrict__ wkT, h16* __restrict__ wvT,
    h16* __restrict__ woT) {
  const int idx = blockIdx.x * 256 + threadIdx.x;  // 0 .. 1,048,575
  const int h   = idx >> 16;
  const int rem = idx & 65535;
  const int d   = rem >> 6;   // 0..1023
  const int c   = rem & 63;   // 0..63
  const int t   = (h << 16) + (c << 10) + d;
  wqT[t] = (h16)Wq[idx];
  wkT[t] = (h16)Wk[idx];
  wvT[t] = (h16)Wv[idx];
  if (idx < D_MODEL * DK) woT[(c << 10) + d] = (h16)Wo[idx];
}

// ---------------------------------------------------------------------------
// Kernel 1: q/k/v = relu(x @ W + b).  q,k stored [bh][s][dk]; v stored
// TRANSPOSED [bh][dk][s] so the P·V fragments in kernel 2 are contiguous.
// x tile staged global->LDS with the Tensor Data Mover (TDM).
// grid = (512 m-tiles, 16 heads), block = 128 (4 waves)
// ---------------------------------------------------------------------------
__global__ __launch_bounds__(128) void qkv_proj_kernel(
    const float* __restrict__ x,
    const h16* __restrict__ wqT, const h16* __restrict__ wkT,
    const h16* __restrict__ wvT,
    const float* __restrict__ bq, const float* __restrict__ bk,
    const float* __restrict__ bv,
    h16* __restrict__ qws, h16* __restrict__ kws, h16* __restrict__ vtws) {
  __shared__ float xstage[16 * D_MODEL];  // 64 KB raw f32 tile (TDM dest)
  __shared__ h16   xtile[16 * D_MODEL];   // 32 KB f16 tile for WMMA A-frags

  const int mt   = blockIdx.x;   // 0..511
  const int h    = blockIdx.y;   // 0..15
  const int tid  = threadIdx.x;
  const int row0 = mt * 16;

  // --- TDM: DMA the 16x1024 f32 tile of x into LDS (wave 0 issues) ---------
  if (tid < 32) {
    const unsigned long long gaddr =
        (unsigned long long)(x + (size_t)row0 * D_MODEL);
    const unsigned int laddr = (unsigned int)(size_t)(&xstage[0]);  // LDS byte addr
    u32x4 g0;
    g0[0] = 1u;                                   // count=1, user descriptor
    g0[1] = laddr;                                // lds_addr
    g0[2] = (unsigned int)gaddr;                  // global_addr[31:0]
    g0[3] = (unsigned int)((gaddr >> 32) & 0x1FFFFFFull) | (2u << 30);  // type=2
    i32x8 g1;
    g1[0] = (int)(2u << 16);        // data_size = 4 bytes
    g1[1] = (int)(1024u << 16);     // tensor_dim0 = 1024 (bits 79:48, lo16)
    g1[2] = (int)(8192u << 16);     // tensor_dim0 hi=0 | tensor_dim1 lo16 = 8192
    g1[3] = (int)(1024u << 16);     // tensor_dim1 hi=0 | tile_dim0 = 1024
    g1[4] = 16;                     // tile_dim1 = 16, tile_dim2 = 0
    g1[5] = 1024;                   // tensor_dim0_stride lo32 = 1024 elements
    g1[6] = 0;                      // dim0_stride hi | dim1_stride lo
    g1[7] = 0;
    const i32x4 gz4 = {0, 0, 0, 0};           // groups 2/3 unused (2-D tensor)
    const i32x8 gz8 = {0, 0, 0, 0, 0, 0, 0, 0};
    __builtin_amdgcn_tensor_load_to_lds(g0, g1, gz4, gz4, gz8, 0);
    __builtin_amdgcn_s_wait_tensorcnt(0);
  }
  __syncthreads();

  // f32 -> f16 once, then all waves consume f16 A-fragments via ds_load_b128
  for (int idx = tid; idx < 16 * D_MODEL; idx += 128)
    xtile[idx] = (h16)xstage[idx];
  __syncthreads();

  const int w    = tid >> 5;
  const int lane = tid & 31;
  const int ncol = w * 16;  // wave w owns columns [16w, 16w+16) of q,k,v

  const h16* WT[3] = {wqT + h * (DK * D_MODEL), wkT + h * (DK * D_MODEL),
                      wvT + h * (DK * D_MODEL)};
  const float* bias[3] = {bq + h * DK, bk + h * DK, bv + h * DK};

  v8f acc[3] = {{}, {}, {}};
  for (int kk = 0; kk < D_MODEL; kk += 32) {
    v16h a = load_afrag(&xtile[kk], D_MODEL);  // shared A from LDS
#pragma unroll
    for (int m3 = 0; m3 < 3; ++m3) {
      v16h b = load_bfrag_T(WT[m3] + ncol * D_MODEL + kk, D_MODEL);
      acc[m3] = wmma_f32(a, b, acc[m3]);
    }
  }

  const int n     = lane & 15;
  const int mbase = (lane >> 4) * 8;
  const int bidx  = row0 >> 10;   // batch
  const int srow0 = row0 & 1023;  // seq pos of tile row 0
  const int bh    = bidx * NHEADS + h;

  // q, k: row-major [bh][s][dk]
  h16* qk[2] = {qws, kws};
#pragma unroll
  for (int m3 = 0; m3 < 2; ++m3) {
    const float bval = bias[m3][ncol + n];
    h16* base = qk[m3] + (size_t)bh * SEQ * DK;
#pragma unroll
    for (int r = 0; r < 8; ++r) {
      float val = acc[m3][r] + bval;
      val = val > 0.f ? val : 0.f;
      base[(srow0 + mbase + r) * DK + ncol + n] = (h16)val;
    }
  }
  // v: transposed [bh][dk][s] (per-lane contiguous 8-half run over r)
  {
    const float bval = bias[2][ncol + n];
    h16* base = vtws + ((size_t)bh * DK + ncol + n) * SEQ + srow0 + mbase;
#pragma unroll
    for (int r = 0; r < 8; ++r) {
      float val = acc[2][r] + bval;
      val = val > 0.f ? val : 0.f;
      base[r] = (h16)val;
    }
  }
}

// ---------------------------------------------------------------------------
// Kernel 2: fused raw-score write (non-temporal) + flash attention.
// One wave per (b,h, 16-query block); 32 keys per iteration.
// grid = 2048, block = 128 (4 waves) -> 8192 waves = 128 bh * 64 m-tiles
// ---------------------------------------------------------------------------
__global__ __launch_bounds__(128) void attn_kernel(
    const h16* __restrict__ qws, const h16* __restrict__ kws,
    const h16* __restrict__ vtws, float* __restrict__ att,
    h16* __restrict__ headsws) {
  __shared__ h16 pbuf[4][16 * 32];  // per-wave prob tile: f32 C -> f16 A via LDS

  const int tid  = threadIdx.x;
  const int w    = tid >> 5;
  const int lane = tid & 31;
  const int wg   = blockIdx.x * 4 + w;  // 0..8191
  const int bh   = wg >> 6;             // 0..127
  const int mt   = wg & 63;             // 0..63

  const h16* qp = qws + (size_t)bh * SEQ * DK;
  const h16* kp = kws + (size_t)bh * SEQ * DK;
  const h16* vp = vtws + (size_t)bh * DK * SEQ;  // [dk][s]
  float* attb   = att + (size_t)bh * SEQ * SEQ;

  const int row0  = mt * 16;
  const int n     = lane & 15;
  const int mbase = (lane >> 4) * 8;

  const v16h q0 = load_afrag(qp + row0 * DK, DK);       // dk 0..31
  const v16h q1 = load_afrag(qp + row0 * DK + 32, DK);  // dk 32..63

  float mrun[8], lrun[8];
#pragma unroll
  for (int r = 0; r < 8; ++r) { mrun[r] = -1e30f; lrun[r] = 0.f; }
  v8f acc[4] = {{}, {}, {}, {}};

  for (int kblk = 0; kblk < SEQ; kblk += 32) {
    v8f st[2];
    float tmax[8];
#pragma unroll
    for (int jt = 0; jt < 2; ++jt) {
      const int nb = kblk + jt * 16;  // 16 key columns
      v16h bk0 = load_bfrag_T(kp + nb * DK, DK);       // dk 0..31
      v16h bk1 = load_bfrag_T(kp + nb * DK + 32, DK);  // dk 32..63
      v8f s = {};
      s = wmma_f32(q0, bk0, s);
      s = wmma_f32(q1, bk1, s);
      // raw (unscaled) scores -> output; streaming store, bypass cache
#pragma unroll
      for (int r = 0; r < 8; ++r)
        __builtin_nontemporal_store(
            s[r], &attb[(size_t)(row0 + mbase + r) * SEQ + nb + n]);
#pragma unroll
      for (int r = 0; r < 8; ++r) s[r] *= 0.125f;  // 1/sqrt(64)
      st[jt] = s;
#pragma unroll
      for (int r = 0; r < 8; ++r) {  // row max within each 16-lane half
        float v = s[r];
        v = fmaxf(v, __shfl_xor(v, 1, 32));
        v = fmaxf(v, __shfl_xor(v, 2, 32));
        v = fmaxf(v, __shfl_xor(v, 4, 32));
        v = fmaxf(v, __shfl_xor(v, 8, 32));
        tmax[r] = (jt == 0) ? v : fmaxf(tmax[r], v);
      }
    }

    float coef[8], mnew[8], ladd[8];
#pragma unroll
    for (int r = 0; r < 8; ++r) {
      mnew[r] = fmaxf(mrun[r], tmax[r]);
      coef[r] = __expf(mrun[r] - mnew[r]);
      mrun[r] = mnew[r];
      ladd[r] = 0.f;
    }
#pragma unroll
    for (int nt = 0; nt < 4; ++nt)
#pragma unroll
      for (int r = 0; r < 8; ++r) acc[nt][r] *= coef[r];

#pragma unroll
    for (int jt = 0; jt < 2; ++jt) {
#pragma unroll
      for (int r = 0; r < 8; ++r) {
        float e = __expf(st[jt][r] - mnew[r]);
        pbuf[w][(mbase + r) * 32 + jt * 16 + n] = (h16)e;  // C-layout -> LDS
        float s = e;
        s += __shfl_xor(s, 1, 32);
        s += __shfl_xor(s, 2, 32);
        s += __shfl_xor(s, 4, 32);
        s += __shfl_xor(s, 8, 32);
        ladd[r] += s;
      }
    }
#pragma unroll
    for (int r = 0; r < 8; ++r) lrun[r] = lrun[r] * coef[r] + ladd[r];

    // same-wave DS ops are in-order: read probs back as an A fragment
    const v16h pf = load_afrag(&pbuf[w][0], 32);
#pragma unroll
    for (int nt = 0; nt < 4; ++nt) {
      // B[k][n] = V[key=kblk+k][col=nt*16+n] = vT[(nt*16+n)*SEQ + kblk+k]
      v16h bv = load_bfrag_T(vp + (nt * 16) * SEQ + kblk, SEQ);
      acc[nt] = wmma_f32(pf, bv, acc[nt]);
    }
  }

  // normalize and store heads as f16 [b*S + s][h*64 + c]
  const int b_ = bh >> 4, h_ = bh & 15;
#pragma unroll
  for (int r = 0; r < 8; ++r) {
    const float inv  = 1.f / lrun[r];
    const int   rowg = b_ * SEQ + row0 + mbase + r;
#pragma unroll
    for (int nt = 0; nt < 4; ++nt) {
      headsws[(size_t)rowg * (NHEADS * DK) + h_ * DK + nt * 16 + n] =
          (h16)(acc[nt][r] * inv);
    }
  }
}

// ---------------------------------------------------------------------------
// Kernel 3: z_out = relu(heads @ Wo + bo)   [8192,1024]x[1024,64]
// grid = 512, block = 128 (wave w owns column tile w)
// ---------------------------------------------------------------------------
__global__ __launch_bounds__(128) void out_proj_kernel(
    const h16* __restrict__ headsws, const h16* __restrict__ woT,
    const float* __restrict__ bo, float* __restrict__ zout) {
  const int tid  = threadIdx.x;
  const int w    = tid >> 5;
  const int lane = tid & 31;
  const int mt   = blockIdx.x;
  const int ncol = w * 16;
  const int row0 = mt * 16;

  v8f acc = {};
  for (int kk = 0; kk < NHEADS * DK; kk += 32) {
    v16h a = load_afrag(headsws + (size_t)row0 * (NHEADS * DK) + kk,
                        NHEADS * DK);
    v16h b = load_bfrag_T(woT + ncol * D_MODEL + kk, D_MODEL);
    acc = wmma_f32(a, b, acc);
  }

  const int n     = lane & 15;
  const int mbase = (lane >> 4) * 8;
  const float bb  = bo[ncol + n];
#pragma unroll
  for (int r = 0; r < 8; ++r) {
    float val = acc[r] + bb;
    val = val > 0.f ? val : 0.f;
    zout[(row0 + mbase + r) * DK + ncol + n] = val;
  }
}

// ---------------------------------------------------------------------------
extern "C" void kernel_launch(void* const* d_in, const int* in_sizes, int n_in,
                              void* d_out, int out_size, void* d_ws,
                              size_t ws_size, hipStream_t stream) {
  (void)in_sizes; (void)n_in; (void)out_size; (void)ws_size;

  const float* x  = (const float*)d_in[0];
  const float* Wq = (const float*)d_in[1];
  const float* bq = (const float*)d_in[2];
  const float* Wk = (const float*)d_in[3];
  const float* bk = (const float*)d_in[4];
  const float* Wv = (const float*)d_in[5];
  const float* bv = (const float*)d_in[6];
  const float* Wo = (const float*)d_in[7];
  const float* bo = (const float*)d_in[8];

  float* zout = (float*)d_out;             // [8192, 64]
  float* att  = zout + (size_t)ROWS * DK;  // [128, 1024, 1024]

  // workspace layout (halves):
  const size_t per  = (size_t)BHN * SEQ * DK;  // 8,388,608
  const size_t wsz  = (size_t)NHEADS * DK * D_MODEL;  // 1,048,576
  h16* qws   = (h16*)d_ws;
  h16* kws   = qws + per;
  h16* vtws  = kws + per;        // V transposed [bh][dk][s]
  h16* heads = vtws + per;
  h16* wqT   = heads + per;      // weights f16 column-major
  h16* wkT   = wqT + wsz;
  h16* wvT   = wkT + wsz;
  h16* woT   = wvT + wsz;        // [64][1024]

  prep_weights_kernel<<<dim3(4096), 256, 0, stream>>>(Wq, Wk, Wv, Wo, wqT, wkT,
                                                      wvT, woT);
  qkv_proj_kernel<<<dim3(512, 16), 128, 0, stream>>>(x, wqT, wkT, wvT, bq, bk,
                                                     bv, qws, kws, vtws);
  attn_kernel<<<dim3(2048), 128, 0, stream>>>(qws, kws, vtws, att, heads);
  out_proj_kernel<<<dim3(512), 128, 0, stream>>>(heads, woT, bo, zout);
}
